// Model_60576218743057
// MI455X (gfx1250) — compile-verified, matrix operand order checked by reading, not run on hardware
//
#include <hip/hip_runtime.h>
#include <hip/hip_bf16.h>

typedef __attribute__((ext_vector_type(16))) _Float16 v16h;
typedef __attribute__((ext_vector_type(8)))  float    v8f;

#define DEVINL __device__ __forceinline__

constexpr int WPB = 4;          // waves per block
constexpr int TPB = WPB * 32;   // 128 threads

// ---------------- shared memory layout ----------------
struct WaveScratch {
    _Float16 x_lp[16][96];   // A: rows 0..9 left feats (K 0..35), row 10 player feats (K 36..71)
    _Float16 x2_lp[16][64];  // LN+relu hidden for [l;p], K-pad 64
    _Float16 x_r[16][64];    // A: right feats
    _Float16 x2_r[16][64];   // LN+relu hidden for r
    _Float16 qa[16][64];     // q1 rows (A for scores)
    _Float16 krb[16][64];    // k1_r rows (Bt for scores: [N=j][K=d])
    _Float16 vrt[48][32];    // v1_r transposed ([N=d][K=j]) for P@V
    _Float16 pa[16][32];     // softmax probs as A ([i][j], K-pad 32)
    _Float16 emb[16][96];    // [l_emb rows 0..9 ; p_emb row 10] = [v1 | att@v1r]
    float    pq2[96];
    float    t[16];
};

struct __align__(16) Smem {
    _Float16 wt_lp[48][96];  // [n][k]: k<36 W_left, 36..71 W_player, pad 0
    _Float16 wt_r [48][64];
    _Float16 wt_q1[48][64];
    _Float16 wt_v1[48][64];
    _Float16 wt_k1[48][64];
    _Float16 wt_k2[96][96];
    _Float16 wq2t [96][96];  // Wq2 transposed [n][k] for VALU dot
    float bL[48], bP[48], bR[48], gm[48], bt[48];
    WaveScratch ws[WPB];
};

// ---------------- helpers ----------------
DEVINL v16h frag_ld(const _Float16* mat, int stride, int row, int kbase, int kh) {
    const _Float16* p = mat + row * stride + kbase + kh * 8;
    union { uint4 u[2]; v16h v; } x;
    x.u[0] = *(const uint4*)p;
    x.u[1] = *(const uint4*)(p + 16);
    return x.v;
}

// acc += A(16 x 32*ksteps) * B(32*ksteps x 16), A row-major [16][as], Bt row-major [N][bs]
DEVINL void gemm(v8f& acc, const _Float16* A, int as,
                 const _Float16* Bt, int bs, int nbase, int ksteps, int lane) {
    const int r  = lane & 15;
    const int kh = lane >> 4;
#pragma unroll
    for (int ks = 0; ks < ksteps; ++ks) {
        v16h a = frag_ld(A,  as, r,          ks * 32, kh);
        v16h b = frag_ld(Bt, bs, nbase + r,  ks * 32, kh);
        acc = __builtin_amdgcn_wmma_f32_16x16x32_f16(false, a, false, b,
                                                     (short)0, acc, false, false);
    }
}

DEVINL float red16(float v) {
    v += __shfl_xor(v, 1, 32);
    v += __shfl_xor(v, 2, 32);
    v += __shfl_xor(v, 4, 32);
    v += __shfl_xor(v, 8, 32);
    return v;
}
DEVINL float rmax16(float v) {
    v = fmaxf(v, __shfl_xor(v, 1, 32));
    v = fmaxf(v, __shfl_xor(v, 2, 32));
    v = fmaxf(v, __shfl_xor(v, 4, 32));
    v = fmaxf(v, __shfl_xor(v, 8, 32));
    return v;
}

DEVINL void ld8(const _Float16* p, float* f) {
    union { uint4 u; _Float16 h[8]; } x;
    x.u = *(const uint4*)p;
#pragma unroll
    for (int i = 0; i < 8; ++i) f[i] = (float)x.h[i];
}

// LayerNorm(+bias)+ReLU epilogue over 3 N-tiles (48 cols), store f16 rows (< nrows)
DEVINL void ln_relu_store(const v8f* acc, const float* bmain, const float* bspec, int specrow,
                          const float* gmv, const float* btv,
                          _Float16 (*dst)[64], int nrows, int lane) {
    const int n = lane & 15, half = lane >> 4;
    float g[3], be[3], bm[3], bs[3];
#pragma unroll
    for (int nt = 0; nt < 3; ++nt) {
        int c = nt * 16 + n;
        g[nt] = gmv[c]; be[nt] = btv[c]; bm[nt] = bmain[c]; bs[nt] = bspec[c];
    }
#pragma unroll
    for (int v = 0; v < 8; ++v) {
        int m = half * 8 + v;
        float x0 = acc[0][v] + (m == specrow ? bs[0] : bm[0]);
        float x1 = acc[1][v] + (m == specrow ? bs[1] : bm[1]);
        float x2 = acc[2][v] + (m == specrow ? bs[2] : bm[2]);
        float s  = red16(x0 + x1 + x2);
        float q  = red16(x0 * x0 + x1 * x1 + x2 * x2);
        float mu  = s * (1.0f / 48.0f);
        float var = q * (1.0f / 48.0f) - mu * mu;
        float rs  = rsqrtf(var + 1e-5f);
        float y0 = fmaxf(0.f, (x0 - mu) * rs * g[0] + be[0]);
        float y1 = fmaxf(0.f, (x1 - mu) * rs * g[1] + be[1]);
        float y2 = fmaxf(0.f, (x2 - mu) * rs * g[2] + be[2]);
        if (m < nrows) {
            dst[m][n]      = (_Float16)y0;
            dst[m][16 + n] = (_Float16)y1;
            dst[m][32 + n] = (_Float16)y2;
        }
    }
}

DEVINL void store_rows64(const v8f& acc, _Float16 (*dst)[64], int nt, int nrows, int lane) {
    const int n = lane & 15, half = lane >> 4;
#pragma unroll
    for (int v = 0; v < 8; ++v) {
        int m = half * 8 + v;
        if (m < nrows) dst[m][nt * 16 + n] = (_Float16)acc[v];
    }
}

// ---------------- fused kernel ----------------
__global__ __launch_bounds__(TPB, 1) void fused_gfootball(
    const float* __restrict__ player, const float* __restrict__ left, const float* __restrict__ right,
    const float* __restrict__ Wp,  const float* __restrict__ bp,
    const float* __restrict__ Wl,  const float* __restrict__ bl,
    const float* __restrict__ Wr,  const float* __restrict__ br,
    const float* __restrict__ gma, const float* __restrict__ bta,
    const float* __restrict__ Wq1, const float* __restrict__ Wv1, const float* __restrict__ Wk1,
    const float* __restrict__ Wq2, const float* __restrict__ Wk2,
    float* __restrict__ out, int B) {

    __shared__ Smem sm;
    const int tid  = threadIdx.x;
    const int lane = tid & 31;
    const int wv   = tid >> 5;

    // zero entire LDS (establishes all pad-zeros once; pads are never overwritten)
    for (int i = tid; i < (int)(sizeof(Smem) / 4); i += TPB) ((float*)&sm)[i] = 0.f;
    __syncthreads();

    // stage weights (transposed, f16) into LDS
    for (int e = tid; e < 48 * 72; e += TPB) {
        int n = e / 72, k = e % 72;
        sm.wt_lp[n][k] = (_Float16)(k < 36 ? Wl[k * 48 + n] : Wp[(k - 36) * 48 + n]);
    }
    for (int e = tid; e < 48 * 36; e += TPB) {
        int n = e / 36, k = e % 36;
        sm.wt_r[n][k] = (_Float16)Wr[k * 48 + n];
    }
    for (int e = tid; e < 48 * 48; e += TPB) {
        int n = e / 48, k = e % 48;
        sm.wt_q1[n][k] = (_Float16)Wq1[k * 48 + n];
        sm.wt_v1[n][k] = (_Float16)Wv1[k * 48 + n];
        sm.wt_k1[n][k] = (_Float16)Wk1[k * 48 + n];
    }
    for (int e = tid; e < 96 * 96; e += TPB) {
        int n = e / 96, k = e % 96;
        sm.wt_k2[n][k] = (_Float16)Wk2[k * 96 + n];
        sm.wq2t[n][k]  = (_Float16)Wq2[k * 96 + n];
    }
    for (int e = tid; e < 48; e += TPB) {
        sm.bL[e] = bl[e]; sm.bP[e] = bp[e]; sm.bR[e] = br[e];
        sm.gm[e] = gma[e]; sm.bt[e] = bta[e];
    }
    __syncthreads();

    WaveScratch& W = sm.ws[wv];
    const int n    = lane & 15;
    const int half = lane >> 4;
    const v8f vz   = {0.f, 0.f, 0.f, 0.f, 0.f, 0.f, 0.f, 0.f};

    for (int b = blockIdx.x * WPB + wv; b < B; b += gridDim.x * WPB) {
        // ---- stage inputs as f16 A-matrices ----
        for (int idx = lane; idx < 360; idx += 32)
            W.x_lp[idx / 36][idx % 36] = (_Float16)left[(size_t)b * 360 + idx];
        for (int idx = lane; idx < 36; idx += 32)
            W.x_lp[10][36 + idx] = (_Float16)player[(size_t)b * 36 + idx];
        for (int idx = lane; idx < 396; idx += 32)
            W.x_r[idx / 36][idx % 36] = (_Float16)right[(size_t)b * 396 + idx];

        // ---- input projections + LN + ReLU ----
        v8f accL[3];
#pragma unroll
        for (int nt = 0; nt < 3; ++nt) {
            accL[nt] = vz;
            gemm(accL[nt], &W.x_lp[0][0], 96, &sm.wt_lp[0][0], 96, nt * 16, 3, lane);
        }
        ln_relu_store(accL, sm.bL, sm.bP, 10, sm.gm, sm.bt, W.x2_lp, 11, lane);

        v8f accR[3];
#pragma unroll
        for (int nt = 0; nt < 3; ++nt) {
            accR[nt] = vz;
            gemm(accR[nt], &W.x_r[0][0], 64, &sm.wt_r[0][0], 64, nt * 16, 2, lane);
        }
        ln_relu_store(accR, sm.bR, sm.bR, -1, sm.gm, sm.bt, W.x2_r, 11, lane);

        // ---- q1 / v1 for [l;p], k1 / v1 for r ----
#pragma unroll
        for (int nt = 0; nt < 3; ++nt) {
            v8f a = vz;
            gemm(a, &W.x2_lp[0][0], 64, &sm.wt_q1[0][0], 64, nt * 16, 2, lane);
            store_rows64(a, W.qa, nt, 11, lane);
        }
#pragma unroll
        for (int nt = 0; nt < 3; ++nt) {
            v8f a = vz;
            gemm(a, &W.x2_lp[0][0], 64, &sm.wt_v1[0][0], 64, nt * 16, 2, lane);
#pragma unroll
            for (int v = 0; v < 8; ++v) {         // v1 -> emb cols 0..47
                int m = half * 8 + v;
                if (m < 11) W.emb[m][nt * 16 + n] = (_Float16)a[v];
            }
        }
#pragma unroll
        for (int nt = 0; nt < 3; ++nt) {
            v8f a = vz;
            gemm(a, &W.x2_r[0][0], 64, &sm.wt_k1[0][0], 64, nt * 16, 2, lane);
            store_rows64(a, W.krb, nt, 11, lane);
        }
#pragma unroll
        for (int nt = 0; nt < 3; ++nt) {
            v8f a = vz;
            gemm(a, &W.x2_r[0][0], 64, &sm.wt_v1[0][0], 64, nt * 16, 2, lane);
#pragma unroll
            for (int v = 0; v < 8; ++v) {         // transposed store for B use
                int m = half * 8 + v;
                if (m < 11) W.vrt[nt * 16 + n][m] = (_Float16)a[v];
            }
        }

        // ---- scores vs right + softmax (rows 0..9 = att_lr, row 10 = att_pr) ----
        {
            v8f sc = vz;
            gemm(sc, &W.qa[0][0], 64, &W.krb[0][0], 64, 0, 2, lane);
            const int j = n;
#pragma unroll
            for (int v = 0; v < 8; ++v) {
                int i = half * 8 + v;
                float x  = (j < 11) ? sc[v] : -3.0e38f;
                float mx = rmax16(x);
                float e  = (j < 11) ? __expf(x - mx) : 0.f;
                float s  = red16(e);
                float p  = e / s;
                if (i < 11 && j < 11) W.pa[i][j] = (_Float16)p;
                if (i < 10 && j < 11)
                    out[(size_t)B * 10 + (size_t)b * 110 + i * 11 + j] = p;
            }
        }

        // ---- O = P @ v1_r  -> emb cols 48..95 ----
#pragma unroll
        for (int nt = 0; nt < 3; ++nt) {
            v8f a = vz;
            gemm(a, &W.pa[0][0], 32, &W.vrt[0][0], 32, nt * 16, 1, lane);
#pragma unroll
            for (int v = 0; v < 8; ++v) {
                int m = half * 8 + v;
                if (m < 11) W.emb[m][48 + nt * 16 + n] = (_Float16)a[v];
            }
        }

        // ---- pq2 = p_emb @ Wq2 (VALU, 3 outputs/lane) ----
        {
            float a0 = 0.f, a1 = 0.f, a2 = 0.f;
#pragma unroll
            for (int kk = 0; kk < 96; kk += 8) {
                float e[8], w0[8], w1[8], w2[8];
                ld8(&W.emb[10][kk], e);
                ld8(&sm.wq2t[lane][kk], w0);
                ld8(&sm.wq2t[lane + 32][kk], w1);
                ld8(&sm.wq2t[lane + 64][kk], w2);
#pragma unroll
                for (int u = 0; u < 8; ++u) {
                    a0 += e[u] * w0[u]; a1 += e[u] * w1[u]; a2 += e[u] * w2[u];
                }
            }
            W.pq2[lane] = a0; W.pq2[lane + 32] = a1; W.pq2[lane + 64] = a2;
        }

        // ---- lk2 = l_emb @ Wk2 fused with dot against pq2 ----
        {
            float tp[8];
#pragma unroll
            for (int v = 0; v < 8; ++v) tp[v] = 0.f;
#pragma unroll
            for (int nt = 0; nt < 6; ++nt) {
                float pq = W.pq2[nt * 16 + n];
                v8f a = vz;
                gemm(a, &W.emb[0][0], 96, &sm.wt_k2[0][0], 96, nt * 16, 3, lane);
#pragma unroll
                for (int v = 0; v < 8; ++v) tp[v] += a[v] * pq;
            }
#pragma unroll
            for (int v = 0; v < 8; ++v) {
                float tv = red16(tp[v]);
                int m = half * 8 + v;
                if (n == 0 && m < 10) W.t[m] = tv;
            }
        }
        asm volatile("s_wait_dscnt 0" ::: "memory");

        // ---- att_pl softmax over 10 ----
        {
            int m2 = n;
            float x  = (m2 < 10) ? W.t[m2] : -3.0e38f;
            float mx = rmax16(x);
            float e  = (m2 < 10) ? __expf(x - mx) : 0.f;
            float s  = red16(e);
            if (lane < 10) out[(size_t)b * 10 + lane] = e / s;
        }
    }
}

extern "C" void kernel_launch(void* const* d_in, const int* in_sizes, int n_in,
                              void* d_out, int out_size, void* d_ws, size_t ws_size,
                              hipStream_t stream) {
    const float* player = (const float*)d_in[0];
    const float* left   = (const float*)d_in[1];
    const float* right  = (const float*)d_in[2];
    const float* Wp  = (const float*)d_in[3];
    const float* bp  = (const float*)d_in[4];
    const float* Wl  = (const float*)d_in[5];
    const float* bl  = (const float*)d_in[6];
    const float* Wr  = (const float*)d_in[7];
    const float* br  = (const float*)d_in[8];
    const float* gm  = (const float*)d_in[9];
    const float* bt  = (const float*)d_in[10];
    const float* Wq1 = (const float*)d_in[11];
    const float* Wv1 = (const float*)d_in[12];
    const float* Wk1 = (const float*)d_in[13];
    const float* Wq2 = (const float*)d_in[14];
    const float* Wk2 = (const float*)d_in[15];
    float* out = (float*)d_out;

    int B = in_sizes[0] / 36;
    int blocks = 4096;
    int waves_needed = (B + WPB - 1) / WPB;
    if (blocks > waves_needed) blocks = waves_needed;

    fused_gfootball<<<dim3(blocks), dim3(TPB), 0, stream>>>(
        player, left, right, Wp, bp, Wl, bl, Wr, br, gm, bt,
        Wq1, Wv1, Wk1, Wq2, Wk2, out, B);
}
